// NeighborAttentionLayer_52553219834577
// MI455X (gfx1250) — matrix operand order repaired and checked by reading, not compiled, not run on hardware
//
#include <hip/hip_runtime.h>
#include <hip/hip_bf16.h>
#include <cstdint>
#include <cmath>

// Problem constants (match reference)
#define B_      64
#define K_      128
#define D_      2560
#define H_      8
#define HD_     320
#define DFF_    1024
#define BIASHID_ 64

typedef __attribute__((ext_vector_type(16))) __bf16 v16bf;
typedef __attribute__((ext_vector_type(8)))  float  v8f;

__device__ __forceinline__ unsigned short f2bf(float f) {
  union { float f; unsigned u; } c; c.f = f;
  unsigned u = c.u;
  unsigned r = u + 0x7FFFu + ((u >> 16) & 1u);   // round-to-nearest-even
  return (unsigned short)(r >> 16);
}

// CDNA5 async global->LDS copy (ASYNCcnt-tracked; no VGPR round trip).
// %0 = VGPR with 32-bit LDS byte offset (VDST field), %1 = 64-bit global address.
__device__ __forceinline__ void async_load_b128(unsigned ldsOff, const void* gaddr) {
  asm volatile("global_load_async_to_lds_b128 %0, %1, off"
               :: "v"(ldsOff), "v"(gaddr)
               : "memory");
}
__device__ __forceinline__ void wait_async0() {
  asm volatile("s_wait_asynccnt 0x0" ::: "memory");
}
__device__ __forceinline__ unsigned lds_off(const void* p) {
  // low 32 bits of a flat LDS-aperture address == group-segment byte offset
  return (unsigned)(size_t)p;
}

// ---------------------------------------------------------------- cvt f32->bf16
__global__ void cvt_f32_bf16(const float* __restrict__ in,
                             unsigned short* __restrict__ out, int n) {
  int i = blockIdx.x * blockDim.x + threadIdx.x;
  if (i < n) out[i] = f2bf(in[i]);
}

// ---------------------------------------------------------------- distance-bias MLP
__global__ void bias_mlp_kernel(const float* __restrict__ dist,
                                const float* __restrict__ w1, const float* __restrict__ b1,
                                const float* __restrict__ w2, const float* __restrict__ b2,
                                float* __restrict__ bph) {
  int i = blockIdx.x * blockDim.x + threadIdx.x;
  if (i >= B_ * K_) return;
  float d = dist[i];
  float acc[H_];
#pragma unroll
  for (int h = 0; h < H_; ++h) acc[h] = b2[h];
  for (int c = 0; c < BIASHID_; ++c) {
    float hm = fmaxf(d * w1[c] + b1[c], 0.f);
#pragma unroll
    for (int h = 0; h < H_; ++h) acc[h] += hm * w2[h * BIASHID_ + c];
  }
#pragma unroll
  for (int h = 0; h < H_; ++h) bph[(size_t)i * H_ + h] = acc[h];
}

// ---------------------------------------------------------------- generic WMMA GEMM
// C[m,n] = sum_k A[m,k] * B[n,k]   (B given as [N,K] row-major, i.e. W for X@W^T)
// Block tile 128x64, k-step 32, 8 waves (4x2), each wave 32x32 via 4 wmma tiles.
// Dense bf16 path uses double-buffered GLOBAL_LOAD_ASYNC_TO_LDS_B128.
struct GemmParams {
  const unsigned short* A;    // bf16 A (when !AF32)
  const float*          Af;   // f32 A (when AF32)
  const unsigned short* Bm;   // bf16 B ([N,K] rm, or [K,N] rm when BTRANS)
  unsigned short*       Cb;   // bf16 out
  float*                Cf;   // f32 out
  long long aSO, aSI, bSO, bSI, cSO, cSI;   // batch strides (outer/inner)
  int lda, ldb, ldc, M, N, Kc, innerB;
  const float* bias;          // per-n bias
  const float* resid; int ldr;        // f32 residual (EPI 3)
  const float* rowBias;               // scores: idx (zo*M+m)*innerB+zi
  const unsigned char* mask;          // scores: idx zo*N+n
  float scale;
};

// EPI: 0=+bias->bf16 (qkv)  1=scale+rowBias+mask->f32 (scores)
//      2=->bf16 (attn@V)    3=+bias+resid->f32 (outproj/ffn2)   4=+bias,relu->bf16 (ffn1)
template <int EPI, bool AF32, bool BTRANS>
__global__ __launch_bounds__(256)
void gemm_wmma(GemmParams p) {
  constexpr bool ASYNC = (!AF32 && !BTRANS);
  constexpr int SA = 128 * 40;   // padded A slab (elements)
  constexpr int SB = 64 * 40;    // padded B slab (elements)
  __shared__ __align__(16) unsigned short sA[(ASYNC ? 2 : 1) * SA];
  __shared__ __align__(16) unsigned short sB[(ASYNC ? 2 : 1) * SB];

  const int tid      = threadIdx.x;
  const int wave     = tid >> 5;
  const int lane     = tid & 31;
  const int lane16   = lane & 15;
  const int laneHalf = lane >> 4;
  const int wm       = wave & 3;     // 4 waves along M
  const int wn       = wave >> 2;    // 2 waves along N
  const int z  = blockIdx.z;
  const int zo = z / p.innerB;
  const int zi = z % p.innerB;
  const int mBase = blockIdx.y * 128;
  const int nBase = blockIdx.x * 64;

  const unsigned short* Ab = AF32 ? nullptr : p.A + zo * p.aSO + zi * p.aSI;
  const float*          Af = AF32 ? p.Af + zo * p.aSO + zi * p.aSI : nullptr;
  const unsigned short* Bb = p.Bm + zo * p.bSO + zi * p.bSI;

  v8f acc[2][2] = {};
  const int kTiles = p.Kc >> 5;

  // fragment builder from an LDS slab pair (per CDNA5 16-bit WMMA layouts)
  auto compute_slab = [&](const unsigned short* bA, const unsigned short* bB) {
    union FragU { v16bf v; uint4 q[2]; } aF[2], bF[2];
#pragma unroll
    for (int t = 0; t < 2; ++t) {
      // A 16x32: lane<16 -> row=lane, K chunks {0..7,16..23}; lane>=16 -> {8..15,24..31}
      int r  = wm * 32 + t * 16 + lane16;
      int kb = laneHalf * 8;
      aF[t].q[0] = *reinterpret_cast<const uint4*>(&bA[r * 40 + kb]);
      aF[t].q[1] = *reinterpret_cast<const uint4*>(&bA[r * 40 + kb + 16]);
      // B 32x16: lane L -> col=L%16, K chunks {0..7,16..23} (L<16) / {8..15,24..31} (L>=16)
      int cN  = wn * 32 + t * 16 + lane16;
      int kb2 = laneHalf * 8;
      bF[t].q[0] = *reinterpret_cast<const uint4*>(&bB[cN * 40 + kb2]);
      bF[t].q[1] = *reinterpret_cast<const uint4*>(&bB[cN * 40 + kb2 + 16]);
    }
#pragma unroll
    for (int tm = 0; tm < 2; ++tm)
#pragma unroll
      for (int tn = 0; tn < 2; ++tn)
        acc[tm][tn] = __builtin_amdgcn_wmma_f32_16x16x32_bf16(
            false, aF[tm].v, false, bF[tn].v, (short)0, acc[tm][tn], false, false);
  };

  if constexpr (ASYNC) {
    // ---------------- double-buffered async pipeline ----------------
    auto issue_slab = [&](int buf, int k0) {
#pragma unroll
      for (int it = 0; it < 2; ++it) {               // A: 128x32 = 512 chunks
        int c = tid + it * 256;
        int row = c >> 2, colc = (c & 3) << 3;
        async_load_b128(lds_off(&sA[buf * SA + row * 40 + colc]),
                        Ab + (size_t)(mBase + row) * p.lda + k0 + colc);
      }
      {                                              // B: 64x32 = 256 chunks
        int row = tid >> 2, colc = (tid & 3) << 3;
        async_load_b128(lds_off(&sB[buf * SB + row * 40 + colc]),
                        Bb + (size_t)(nBase + row) * p.ldb + k0 + colc);
      }
    };

    issue_slab(0, 0);
    for (int kt = 0; kt < kTiles; ++kt) {
      wait_async0();        // this wave's copies into buf kt&1 are done
      __syncthreads();      // everyone's copies are done
      if (kt + 1 < kTiles)  // overlap next slab's copies with compute
        issue_slab((kt + 1) & 1, (kt + 1) << 5);
      compute_slab(&sA[(kt & 1) * SA], &sB[(kt & 1) * SB]);
    }
  } else {
    // ---------------- transform-on-load path (P@V) ----------------
    for (int kt = 0; kt < kTiles; ++kt) {
      const int k0 = kt << 5;
      if constexpr (!AF32) {
#pragma unroll
        for (int it = 0; it < 2; ++it) {
          int c = tid + it * 256;
          int row = c >> 2, colc = (c & 3) << 3;
          uint4 v = *reinterpret_cast<const uint4*>(
              Ab + (size_t)(mBase + row) * p.lda + k0 + colc);
          *reinterpret_cast<uint4*>(&sA[row * 40 + colc]) = v;
        }
      } else {
#pragma unroll
        for (int it = 0; it < 2; ++it) {
          int c = tid + it * 256;
          int row = c >> 2, colc = (c & 3) << 3;
          const float* src = Af + (size_t)(mBase + row) * p.lda + k0 + colc;
          float4 f0 = *reinterpret_cast<const float4*>(src);
          float4 f1 = *reinterpret_cast<const float4*>(src + 4);
          unsigned short tmp[8] = {f2bf(f0.x), f2bf(f0.y), f2bf(f0.z), f2bf(f0.w),
                                   f2bf(f1.x), f2bf(f1.y), f2bf(f1.z), f2bf(f1.w)};
          *reinterpret_cast<uint4*>(&sA[row * 40 + colc]) =
              *reinterpret_cast<uint4*>(tmp);
        }
      }
      if constexpr (!BTRANS) {
        int row = tid >> 2, colc = (tid & 3) << 3;
        uint4 v = *reinterpret_cast<const uint4*>(
            Bb + (size_t)(nBase + row) * p.ldb + k0 + colc);
        *reinterpret_cast<uint4*>(&sB[row * 40 + colc]) = v;
      } else {
        // source is [K,N] row-major (e.g. V[j,d]); transpose while staging
        int k = tid >> 3, cc = (tid & 7) << 3;
        union { uint4 q; unsigned short s[8]; } u;
        u.q = *reinterpret_cast<const uint4*>(
            Bb + (size_t)(k0 + k) * p.ldb + nBase + cc);
#pragma unroll
        for (int e = 0; e < 8; ++e) sB[(cc + e) * 40 + k] = u.s[e];
      }
      __syncthreads();
      compute_slab(sA, sB);
      __syncthreads();
    }
  }

  // ---- epilogue: C/D layout: lane L -> N=L%16, M = vgpr + (L/16)*8
  unsigned short* Cb = p.Cb ? p.Cb + zo * p.cSO + zi * p.cSI : nullptr;
  float*          Cf = p.Cf ? p.Cf + zo * p.cSO + zi * p.cSI : nullptr;
#pragma unroll
  for (int tm = 0; tm < 2; ++tm) {
#pragma unroll
    for (int tn = 0; tn < 2; ++tn) {
#pragma unroll
      for (int v = 0; v < 8; ++v) {
        int m = mBase + wm * 32 + tm * 16 + laneHalf * 8 + v;
        int n = nBase + wn * 32 + tn * 16 + lane16;
        if (m >= p.M || n >= p.N) continue;
        float val = acc[tm][tn][v];
        if constexpr (EPI == 0) {
          val += p.bias[n];
          Cb[(size_t)m * p.ldc + n] = f2bf(val);
        } else if constexpr (EPI == 1) {
          val = val * p.scale + p.rowBias[(size_t)(zo * p.M + m) * p.innerB + zi];
          if (!p.mask[(size_t)zo * p.N + n]) val = -10000.0f;
          Cf[(size_t)m * p.ldc + n] = val;
        } else if constexpr (EPI == 2) {
          Cb[(size_t)m * p.ldc + n] = f2bf(val);
        } else if constexpr (EPI == 3) {
          val += p.bias[n] + p.resid[(size_t)m * p.ldr + n];
          Cf[(size_t)m * p.ldc + n] = val;
        } else if constexpr (EPI == 4) {
          val = fmaxf(val + p.bias[n], 0.f);
          Cb[(size_t)m * p.ldc + n] = f2bf(val);
        }
      }
    }
  }
}

// ---------------------------------------------------------------- softmax over 128-wide rows
__global__ __launch_bounds__(128)
void softmax128(float* __restrict__ s) {
  __shared__ float red[128];
  float* p = s + (size_t)blockIdx.x * 128;
  int t = threadIdx.x;
  float v = p[t];
  red[t] = v; __syncthreads();
  for (int off = 64; off > 0; off >>= 1) {
    if (t < off) red[t] = fmaxf(red[t], red[t + off]);
    __syncthreads();
  }
  float mx = red[0]; __syncthreads();
  float e = __expf(v - mx);
  red[t] = e; __syncthreads();
  for (int off = 64; off > 0; off >>= 1) {
    if (t < off) red[t] += red[t + off];
    __syncthreads();
  }
  p[t] = e * (1.0f / red[0]);
}

// ---------------------------------------------------------------- row LayerNorm over D=2560
template <bool EMIT_BF16>
__global__ __launch_bounds__(256)
void layernorm_rows(const float* __restrict__ in, const float* __restrict__ g,
                    const float* __restrict__ bb, float* __restrict__ outF,
                    unsigned short* __restrict__ outB) {
  __shared__ float red[256];
  const float* x = in + (size_t)blockIdx.x * D_;
  int t = threadIdx.x;
  float s = 0.f, s2 = 0.f;
  for (int i = t; i < D_; i += 256) { float v = x[i]; s += v; s2 += v * v; }
  red[t] = s; __syncthreads();
  for (int off = 128; off > 0; off >>= 1) { if (t < off) red[t] += red[t + off]; __syncthreads(); }
  float mean = red[0] * (1.0f / D_); __syncthreads();
  red[t] = s2; __syncthreads();
  for (int off = 128; off > 0; off >>= 1) { if (t < off) red[t] += red[t + off]; __syncthreads(); }
  float var  = red[0] * (1.0f / D_) - mean * mean;
  float rstd = rsqrtf(var + 1e-5f);
  for (int i = t; i < D_; i += 256) {
    float y = (x[i] - mean) * rstd * g[i] + bb[i];
    outF[(size_t)blockIdx.x * D_ + i] = y;
    if constexpr (EMIT_BF16) outB[(size_t)blockIdx.x * D_ + i] = f2bf(y);
  }
}

// ---------------------------------------------------------------- launch
extern "C" void kernel_launch(void* const* d_in, const int* in_sizes, int n_in,
                              void* d_out, int out_size, void* d_ws, size_t ws_size,
                              hipStream_t stream) {
  const float*         x     = (const float*)d_in[0];
  const float*         dist  = (const float*)d_in[1];
  const unsigned char* mask  = (const unsigned char*)d_in[2];
  const float* qkv_w = (const float*)d_in[3];
  const float* qkv_b = (const float*)d_in[4];
  const float* out_w = (const float*)d_in[5];
  const float* out_b = (const float*)d_in[6];
  const float* bw1   = (const float*)d_in[7];
  const float* bb1   = (const float*)d_in[8];
  const float* bw2   = (const float*)d_in[9];
  const float* bb2   = (const float*)d_in[10];
  const float* fw1   = (const float*)d_in[11];
  const float* fb1   = (const float*)d_in[12];
  const float* fw2   = (const float*)d_in[13];
  const float* fb2   = (const float*)d_in[14];
  const float* ln1g  = (const float*)d_in[15];
  const float* ln1b  = (const float*)d_in[16];
  const float* ln2g  = (const float*)d_in[17];
  const float* ln2b  = (const float*)d_in[18];
  float* outp = (float*)d_out;

  const size_t BK  = (size_t)B_ * K_;          // 8192 rows
  char* ws = (char*)d_ws;
  size_t off = 0;
  auto alloc = [&](size_t bytes) { char* r = ws + off; off += (bytes + 255) & ~(size_t)255; return r; };

  unsigned short* xbf    = (unsigned short*)alloc(BK * D_ * 2);
  unsigned short* qkvwbf = (unsigned short*)alloc((size_t)3 * D_ * D_ * 2);
  unsigned short* outwbf = (unsigned short*)alloc((size_t)D_ * D_ * 2);
  unsigned short* f1wbf  = (unsigned short*)alloc((size_t)DFF_ * D_ * 2);
  unsigned short* f2wbf  = (unsigned short*)alloc((size_t)D_ * DFF_ * 2);
  unsigned short* qkvbf  = (unsigned short*)alloc(BK * 3 * D_ * 2);
  float*          bph    = (float*)alloc(BK * H_ * 4);
  float*          scores = (float*)alloc((size_t)B_ * H_ * K_ * K_ * 4);
  unsigned short* attno  = (unsigned short*)alloc(BK * D_ * 2);
  float*          ybuf   = (float*)alloc(BK * D_ * 4);   // pre-LN1, reused pre-LN2
  float*          x1f    = (float*)alloc(BK * D_ * 4);
  unsigned short* x1bf   = (unsigned short*)alloc(BK * D_ * 2);
  unsigned short* ffnh   = (unsigned short*)alloc(BK * DFF_ * 2);

  // 1) bf16 conversions
  {
    int n;
    n = (int)(BK * D_);        cvt_f32_bf16<<<(n + 255) / 256, 256, 0, stream>>>(x, xbf, n);
    n = 3 * D_ * D_;           cvt_f32_bf16<<<(n + 255) / 256, 256, 0, stream>>>(qkv_w, qkvwbf, n);
    n = D_ * D_;               cvt_f32_bf16<<<(n + 255) / 256, 256, 0, stream>>>(out_w, outwbf, n);
    n = DFF_ * D_;             cvt_f32_bf16<<<(n + 255) / 256, 256, 0, stream>>>(fw1, f1wbf, n);
    n = D_ * DFF_;             cvt_f32_bf16<<<(n + 255) / 256, 256, 0, stream>>>(fw2, f2wbf, n);
  }

  // 2) distance-bias MLP
  bias_mlp_kernel<<<(int)((BK + 255) / 256), 256, 0, stream>>>(dist, bw1, bb1, bw2, bb2, bph);

  // 3) QKV projection: [8192,2560] @ [7680,2560]^T -> bf16 [8192,7680] (+bias)
  {
    GemmParams p{};
    p.A = xbf; p.Bm = qkvwbf; p.Cb = qkvbf;
    p.lda = D_; p.ldb = D_; p.ldc = 3 * D_;
    p.M = (int)BK; p.N = 3 * D_; p.Kc = D_; p.innerB = 1;
    p.bias = qkv_b;
    gemm_wmma<0, false, false><<<dim3((3 * D_) / 64, BK / 128, 1), 256, 0, stream>>>(p);
  }

  // 4) scores = q@k^T/sqrt(HD) + bph + mask, batched over (b,h)
  {
    GemmParams p{};
    p.A  = qkvbf;            p.aSO = (long long)K_ * 3 * D_; p.aSI = HD_;
    p.Bm = qkvbf + D_;       p.bSO = (long long)K_ * 3 * D_; p.bSI = HD_;
    p.Cf = scores;           p.cSO = (long long)H_ * K_ * K_; p.cSI = (long long)K_ * K_;
    p.lda = 3 * D_; p.ldb = 3 * D_; p.ldc = K_;
    p.M = K_; p.N = K_; p.Kc = HD_; p.innerB = H_;
    p.rowBias = bph; p.mask = mask;
    p.scale = 1.0f / sqrtf((float)HD_);
    gemm_wmma<1, false, false><<<dim3(K_ / 64, 1, B_ * H_), 256, 0, stream>>>(p);
  }

  // 5) softmax over keys
  softmax128<<<B_ * H_ * K_, 128, 0, stream>>>(scores);

  // 6) attn_out = P @ V (A is f32 probs, B = V via transposed staging)
  {
    GemmParams p{};
    p.Af = scores;           p.aSO = (long long)H_ * K_ * K_; p.aSI = (long long)K_ * K_;
    p.Bm = qkvbf + 2 * D_;   p.bSO = (long long)K_ * 3 * D_;  p.bSI = HD_;
    p.Cb = attno;            p.cSO = (long long)K_ * D_;      p.cSI = HD_;
    p.lda = K_; p.ldb = 3 * D_; p.ldc = D_;
    p.M = K_; p.N = HD_; p.Kc = K_; p.innerB = H_;
    gemm_wmma<2, true, true><<<dim3(HD_ / 64, 1, B_ * H_), 256, 0, stream>>>(p);
  }

  // 7) out projection + bias + residual(x) -> f32 pre-LN1
  {
    GemmParams p{};
    p.A = attno; p.Bm = outwbf; p.Cf = ybuf;
    p.lda = D_; p.ldb = D_; p.ldc = D_;
    p.M = (int)BK; p.N = D_; p.Kc = D_; p.innerB = 1;
    p.bias = out_b; p.resid = x; p.ldr = D_;
    gemm_wmma<3, false, false><<<dim3(D_ / 64, BK / 128, 1), 256, 0, stream>>>(p);
  }

  // 8) LN1 -> x1 (f32 + bf16)
  layernorm_rows<true><<<(int)BK, 256, 0, stream>>>(ybuf, ln1g, ln1b, x1f, x1bf);

  // 9) FFN1: relu(x1 @ W1^T + b1) -> bf16
  {
    GemmParams p{};
    p.A = x1bf; p.Bm = f1wbf; p.Cb = ffnh;
    p.lda = D_; p.ldb = D_; p.ldc = DFF_;
    p.M = (int)BK; p.N = DFF_; p.Kc = D_; p.innerB = 1;
    p.bias = fb1;
    gemm_wmma<4, false, false><<<dim3(DFF_ / 64, BK / 128, 1), 256, 0, stream>>>(p);
  }

  // 10) FFN2 + bias + residual(x1) -> f32 pre-LN2 (reuse ybuf)
  {
    GemmParams p{};
    p.A = ffnh; p.Bm = f2wbf; p.Cf = ybuf;
    p.lda = DFF_; p.ldb = DFF_; p.ldc = D_;
    p.M = (int)BK; p.N = D_; p.Kc = DFF_; p.innerB = 1;
    p.bias = fb2; p.resid = x1f; p.ldr = D_;
    gemm_wmma<3, false, false><<<dim3(D_ / 64, BK / 128, 1), 256, 0, stream>>>(p);
  }

  // 11) LN2 -> output
  layernorm_rows<false><<<(int)BK, 256, 0, stream>>>(ybuf, ln2g, ln2b, outp, nullptr);
}
